// LrlcClf_57501022159486
// MI455X (gfx1250) — matrix operand order repaired and checked by reading, not compiled, not run on hardware
//
#include <hip/hip_runtime.h>
#include <hip/hip_bf16.h>

typedef _Float16 f16;
typedef __attribute__((ext_vector_type(16))) _Float16 v16h;
typedef __attribute__((ext_vector_type(8)))  float    v8f;
typedef __attribute__((ext_vector_type(4)))  unsigned int u32x4;
typedef __attribute__((ext_vector_type(4)))  int          i32x4;
typedef __attribute__((ext_vector_type(8)))  int          i32x8;

#define EPS_BN 1e-5f

// Detect the CDNA5 Tensor Data Mover builtin (arity differs across toolchains:
// ROCm 7.2 / clang-22 -> 5 args, amdgpu-toolchain / clang-23 -> 6 args).
#if defined(__has_builtin)
#if __has_builtin(__builtin_amdgcn_tensor_load_to_lds)
#define HAVE_TDM 1
#endif
#endif
#ifndef HAVE_TDM
#define HAVE_TDM 0
#endif

#if HAVE_TDM
// Issue a TDM 2D tile load: tile (tile_d0 elems x tile_d1 rows) of 2-byte
// elements from a row-major tensor (row stride = d0_stride elems) into LDS,
// with hardware row padding: after every 16 DWORDs (64B = one 32-elem row)
// insert 2 DWORDs (8B) -> padded LDS row stride of 72B == 36 f16 elements.
__device__ __forceinline__ void tdm_load_2d_to_lds(
    unsigned lds_addr, const void* gaddr,
    unsigned tile_d0, unsigned tile_d1,
    unsigned tensor_d0, unsigned tensor_d1, unsigned d0_stride) {
    unsigned long long ga = (unsigned long long)(uintptr_t)gaddr;
    u32x4 g0;
    g0.x = 1u;                                          // count=1 (valid user D#)
    g0.y = lds_addr;                                    // lds_addr [63:32]
    g0.z = (unsigned)(ga & 0xffffffffu);                // global_addr lo
    g0.w = (unsigned)((ga >> 32) & 0x01ffffffu)         // global_addr [56:32]
           | (2u << 30);                                // type = 2 ("image")
    i32x8 g1;
    g1[0] = (int)((1u << 16)      // data_size = 1 -> 2-byte elements
                  | (1u << 20)    // pad_enable
                  | (3u << 22)    // pad_interval = 16 DWORDs
                  | (1u << 25));  // pad_amount   = 2 DWORDs
    g1[1] = (int)((tensor_d0 & 0xffffu) << 16);                         // dim0[15:0]
    g1[2] = (int)(((tensor_d0 >> 16) & 0xffffu) |
                  ((tensor_d1 & 0xffffu) << 16));                       // dim0[31:16] | dim1[15:0]
    g1[3] = (int)(((tensor_d1 >> 16) & 0xffffu) |
                  ((tile_d0 & 0xffffu) << 16));                         // dim1[31:16] | tile_dim0
    g1[4] = (int)(tile_d1 & 0xffffu);                                   // tile_dim1 (tile_dim2 = 0)
    g1[5] = (int)d0_stride;                                             // dim0_stride lo32
    g1[6] = 0;                                                          // dim0_stride hi | dim1_stride lo
    g1[7] = 0;
    i32x4 g2 = {0, 0, 0, 0};
    i32x4 g3 = {0, 0, 0, 0};
#if __clang_major__ >= 23
    i32x8 g4 = {0, 0, 0, 0, 0, 0, 0, 0};
    __builtin_amdgcn_tensor_load_to_lds(g0, g1, g2, g3, g4, 0);
#else
    __builtin_amdgcn_tensor_load_to_lds(g0, g1, g2, g3, 0);
#endif
}

__device__ __forceinline__ void tdm_wait0() {
#if __has_builtin(__builtin_amdgcn_s_wait_tensorcnt)
    __builtin_amdgcn_s_wait_tensorcnt(0);
#else
    asm volatile("s_wait_tensorcnt 0x0" ::: "memory");
#endif
}
#endif  // HAVE_TDM

// ---------------------------------------------------------------------------
// WMMA fragment loaders (wave32, per CDNA5 ISA 7.12.2 layouts).
// Rows are K-contiguous f16 with even element offsets (4B-aligned pair reads).
// A frag (16x32 f16): lane 0-15 => M=lane, K = {0..7,16..23}; lanes 16-31 add +8 to K.
// B frag (32x16 f16): lane 0-15 => N=lane, K = {0..7,16..23}; lanes 16-31 add +8 to K.
// ---------------------------------------------------------------------------
__device__ __forceinline__ v16h frag_a_row(const f16* rowPtr, int lane) {
    union { unsigned int u[8]; v16h h; } t;
    const unsigned int* p = (const unsigned int*)rowPtr;
    const int kh = (lane >> 4) & 1;
#pragma unroll
    for (int v = 0; v < 4; ++v) {
        t.u[v]     = p[kh * 4 + v];        // K = kh*8 + 2v
        t.u[v + 4] = p[8 + kh * 4 + v];    // K = 16 + kh*8 + 2v
    }
    return t.h;
}

__device__ __forceinline__ v16h frag_b_row(const f16* rowPtr, int lane) {
    union { unsigned int u[8]; v16h h; } t;
    const unsigned int* p = (const unsigned int*)rowPtr;
    const int kh = (lane >> 4) & 1;
#pragma unroll
    for (int v = 0; v < 4; ++v) {
        t.u[v]     = p[kh * 4 + v];        // K = kh*8 + 2v
        t.u[v + 4] = p[8 + kh * 4 + v];    // K = 16 + kh*8 + 2v
    }
    return t.h;
}

// ---------------------------------------------------------------------------
// Small elementwise / setup kernels
// ---------------------------------------------------------------------------
__global__ void fold_bn_kernel(const float* __restrict__ cb, const float* __restrict__ g,
                               const float* __restrict__ bb, const float* __restrict__ m,
                               const float* __restrict__ v, float* __restrict__ scale,
                               float* __restrict__ bias, int C) {
    int c = blockIdx.x * blockDim.x + threadIdx.x;
    if (c < C) {
        float s = g[c] * rsqrtf(v[c] + EPS_BN);
        scale[c] = s;
        bias[c]  = s * cb[c] + bb[c] - m[c] * s;
    }
}

__global__ void f32_to_f16_kernel(const float* __restrict__ in, f16* __restrict__ out, size_t n) {
    size_t i = (size_t)blockIdx.x * 256 + threadIdx.x;
    if (i < n) out[i] = (f16)in[i];
}

__global__ void zero_f16_kernel(f16* __restrict__ p, int n) {
    int i = blockIdx.x * 256 + threadIdx.x;
    if (i < n) p[i] = (f16)0.f;
}

// conv1: Cin=1, direct fp32 conv fused with BN+ReLU+maxpool(2,2).
// out: (B=64, 32, 64, 64) f16
__global__ __launch_bounds__(256) void conv1_fused_kernel(
    const float* __restrict__ x, const float* __restrict__ w,
    const float* __restrict__ scale, const float* __restrict__ bias,
    f16* __restrict__ out) {
    __shared__ float wsh[32 * 9];
    for (int i = threadIdx.x; i < 32 * 9; i += 256) wsh[i] = w[i];
    __syncthreads();
    size_t gid = (size_t)blockIdx.x * 256 + threadIdx.x;
    if (gid >= (size_t)64 * 32 * 64 * 64) return;
    int px = gid & 63;
    int py = (gid >> 6) & 63;
    int co = (gid >> 12) & 31;
    int b  = (int)(gid >> 17);
    const float* wk = &wsh[co * 9];
    const float* xb = x + (size_t)b * 128 * 128;
    float s = scale[co], bi = bias[co];
    float mx = -1e30f;
#pragma unroll
    for (int dy = 0; dy < 2; ++dy) {
#pragma unroll
        for (int dx = 0; dx < 2; ++dx) {
            int oy = py * 2 + dy, ox = px * 2 + dx;
            float a = 0.f;
#pragma unroll
            for (int kh = 0; kh < 3; ++kh) {
#pragma unroll
                for (int kw = 0; kw < 3; ++kw) {
                    int iy = oy + kh - 1, ix = ox + kw - 1;
                    if (iy >= 0 && iy < 128 && ix >= 0 && ix < 128)
                        a += xb[iy * 128 + ix] * wk[kh * 3 + kw];
                }
            }
            float r = a * s + bi;
            r = r > 0.f ? r : 0.f;
            mx = r > mx ? r : mx;
        }
    }
    out[gid] = (f16)mx;
}

// maxpool 2x2 on f16 NCHW (flattened BC planes)
__global__ void pool2x2_kernel(const f16* __restrict__ in, f16* __restrict__ out,
                               int H, int W, int total) {
    int idx = blockIdx.x * 256 + threadIdx.x;
    if (idx >= total) return;
    int W2 = W >> 1, H2 = H >> 1;
    int px = idx % W2;
    int t  = idx / W2;
    int py = t % H2;
    int bc = t / H2;
    const f16* p = in + ((size_t)bc * H + 2 * py) * W + 2 * px;
    float a = (float)p[0], b = (float)p[1], c = (float)p[W], d = (float)p[W + 1];
    out[idx] = (f16)fmaxf(fmaxf(a, b), fmaxf(c, d));
}

// combining weights: softmax over rank (R=4) at each of 16x16 locations
__global__ void cw_softmax_kernel(const float* __restrict__ cwh, const float* __restrict__ cww,
                                  float* __restrict__ cw) {
    int i = threadIdx.x;                 // 256 locations
    int h = i >> 4, w = i & 15;
    float l[4];
    float mx = -1e30f;
#pragma unroll
    for (int r = 0; r < 4; ++r) {
        l[r] = cwh[h * 4 + r] + cww[w * 4 + r];
        mx = l[r] > mx ? l[r] : mx;
    }
    float sum = 0.f;
#pragma unroll
    for (int r = 0; r < 4; ++r) { l[r] = expf(l[r] - mx); sum += l[r]; }
    float inv = 1.f / sum;
#pragma unroll
    for (int r = 0; r < 4; ++r) cw[i * 4 + r] = l[r] * inv;
}

// LRLC rank combine + low-rank bias + ReLU; writes f16 FC1 input (B,32768)
__global__ void lrlc_combine_kernel(const float* __restrict__ y,       // (B,512,256)
                                    const float* __restrict__ cw,      // (256,4)
                                    const float* __restrict__ bb,      // (4,128)
                                    f16* __restrict__ out) {           // (B, 128*256)
    int idx = blockIdx.x * 256 + threadIdx.x;  // 64*128*256
    if (idx >= 64 * 128 * 256) return;
    int hw = idx & 255;
    int t  = idx >> 8;
    int c  = t & 127;
    int b  = t >> 7;
    const float* w4 = cw + hw * 4;
    float a = 0.f;
#pragma unroll
    for (int r = 0; r < 4; ++r)
        a += w4[r] * (y[(((size_t)b * 512) + r * 128 + c) * 256 + hw] + bb[r * 128 + c]);
    a = a > 0.f ? a : 0.f;
    out[(size_t)b * 32768 + c * 256 + hw] = (f16)a;
}

// ---------------------------------------------------------------------------
// Implicit-GEMM conv via WMMA: M=COUT, N=pixels, K=CIN*9 (multiple of 32).
// Block = 256 threads (8 waves); covers 32 pixels x all COUT.
// Wave w: n-subtile = w&1, m-group = w>>1 (m tiles strided by 4).
// Weight tiles are staged into LDS by the Tensor Data Mover (one DMA per
// k-chunk, issued by wave 0, overlapped with the im2col staging) when the
// TDM builtin is available; otherwise by wide uint2 copies.
// mode 0: y = relu(scale[m]*acc + bias[m]) -> f16; mode 1: raw f32 out.
// ---------------------------------------------------------------------------
template <int CIN, int COUT, int HH, int WW>
__global__ __launch_bounds__(256) void conv_wmma_kernel(
    const f16* __restrict__ act, const f16* __restrict__ w16,
    const float* __restrict__ scale, const float* __restrict__ bias,
    f16* __restrict__ out16, float* __restrict__ out32, int mode) {
    constexpr int K      = CIN * 9;
    constexpr int KSTEPS = K / 32;
    constexpr int MT     = COUT / 16;
    constexpr int NACC   = (MT + 3) / 4;
    constexpr int LDK    = 36;   // padded row stride (f16 elems); 72B = 64B row + 8B pad

    __shared__ f16 lA[COUT * LDK];   // weights tile: COUT x 32 (padded rows)
    __shared__ f16 lB[32 * LDK];     // im2col tile : 32 pixels x 32 K

    const int tid  = threadIdx.x;
    const int lane = tid & 31;
    const int wave = tid >> 5;
    const int nt   = wave & 1;
    const int mg   = wave >> 1;
    const int b    = blockIdx.y;
    const int pixBase = blockIdx.x * 32;

    v8f zero = {};
    v8f acc[NACC];
#pragma unroll
    for (int i = 0; i < NACC; ++i) acc[i] = zero;

    for (int kc = 0; kc < KSTEPS; ++kc) {
        const int kOff = kc * 32;
#if HAVE_TDM
        // Kick off the weight-tile DMA first so it overlaps im2col staging.
        if (wave == 0) {
            tdm_load_2d_to_lds((unsigned)(uintptr_t)&lA[0], w16 + kOff,
                               /*tile*/ 32u, (unsigned)COUT,
                               /*tensor*/ (unsigned)K, (unsigned)COUT,
                               /*stride*/ (unsigned)K);
        }
#else
        // Fallback: wide 8-byte copies (both sides 8B-aligned).
        for (int idx = tid; idx < COUT * 8; idx += 256) {
            int m = idx >> 3, c = idx & 7;
            *(uint2*)&lA[m * LDK + c * 4] =
                *(const uint2*)(w16 + (size_t)m * K + kOff + c * 4);
        }
#endif
        // stage im2col patch: 32 pixels x 32 K (scattered gather)
        for (int idx = tid; idx < 32 * 32; idx += 256) {
            int pp = idx >> 5, kk = idx & 31;
            int k  = kOff + kk;
            int ci = k / 9;
            int r  = k - ci * 9;
            int kh = r / 3, kw = r - kh * 3;
            int p  = pixBase + pp;
            int y  = p / WW, xq = p - y * WW;
            int iy = y + kh - 1, ix = xq + kw - 1;
            f16 vv = (f16)0.f;
            if (iy >= 0 && iy < HH && ix >= 0 && ix < WW)
                vv = act[((size_t)b * CIN + ci) * (HH * WW) + iy * WW + ix];
            lB[pp * LDK + kk] = vv;
        }
#if HAVE_TDM
        if (wave == 0) tdm_wait0();   // wave 0 arrives at the barrier only once DMA landed
#endif
        __syncthreads();
        v16h bf = frag_b_row(&lB[(nt * 16 + (lane & 15)) * LDK], lane);
#pragma unroll
        for (int a = 0; a < NACC; ++a) {
            int mt = mg + a * 4;          // wave-uniform predicate: EXEC stays all-ones
            if (mt < MT) {
                v16h af = frag_a_row(&lA[(mt * 16 + (lane & 15)) * LDK], lane);
                acc[a] = __builtin_amdgcn_wmma_f32_16x16x32_f16(
                    false, af, false, bf, (short)0, acc[a], false, false);
            }
        }
        __syncthreads();
    }

    const int nloc = nt * 16 + (lane & 15);
    const int p    = pixBase + nloc;
    const int kh2  = lane >> 4;
#pragma unroll
    for (int a = 0; a < NACC; ++a) {
        int mt = mg + a * 4;
        if (mt < MT) {
#pragma unroll
            for (int v = 0; v < 8; ++v) {
                int m = mt * 16 + v + 8 * kh2;
                float val = acc[a][v];
                if (mode == 0) {
                    val = val * scale[m] + bias[m];
                    val = val > 0.f ? val : 0.f;
                    out16[((size_t)b * COUT + m) * (HH * WW) + p] = (f16)val;
                } else {
                    out32[((size_t)b * COUT + m) * (HH * WW) + p] = val;
                }
            }
        }
    }
}

// ---------------------------------------------------------------------------
// FC GEMM via WMMA: D = A(MxK) * B(NxK)^T + bias; one wave per 16x16 tile.
// A, B rows are K-contiguous f16. M must be a multiple of 16 (A zero-padded);
// only rows < Mvalid are stored.
// ---------------------------------------------------------------------------
__global__ __launch_bounds__(256) void fc_wmma_kernel(
    const f16* __restrict__ A, const f16* __restrict__ Bm,
    const float* __restrict__ bias,
    f16* __restrict__ out16, float* __restrict__ out32,
    int M, int Mvalid, int N, int K, int relu, int ldo) {
    const int tid  = threadIdx.x;
    const int lane = tid & 31;
    const int wave = tid >> 5;
    const int nTiles = N >> 4;
    const int tileId = blockIdx.x * 8 + wave;
    const int mt = tileId / nTiles;
    const int nt = tileId - mt * nTiles;
    if (mt * 16 >= M) return;                 // wave-uniform exit
    const int rowA = mt * 16 + (lane & 15);
    const int rowB = nt * 16 + (lane & 15);
    const int kh   = lane >> 4;
    const f16* pA = A + (size_t)rowA * K;
    const f16* pB = Bm + (size_t)rowB * K;
    v8f acc = {};
    for (int kc = 0; kc < K; kc += 32) {
        if (kc + 32 < K) __builtin_prefetch(pA + kc + 32, 0, 1);
        v16h af = frag_a_row(pA + kc, lane);
        v16h bf = frag_b_row(pB + kc, lane);
        acc = __builtin_amdgcn_wmma_f32_16x16x32_f16(
            false, af, false, bf, (short)0, acc, false, false);
    }
    const int n = rowB;
#pragma unroll
    for (int v = 0; v < 8; ++v) {
        int m = mt * 16 + v + 8 * kh;
        if (m < Mvalid) {
            float val = acc[v] + (bias ? bias[m] : 0.f);
            if (relu) val = val > 0.f ? val : 0.f;
            if (out16) out16[(size_t)n * ldo + m] = (f16)val;
            else       out32[(size_t)n * ldo + m] = val;
        }
    }
}

// ---------------------------------------------------------------------------
// Host orchestration
// ---------------------------------------------------------------------------
extern "C" void kernel_launch(void* const* d_in, const int* in_sizes, int n_in,
                              void* d_out, int out_size, void* d_ws, size_t ws_size,
                              hipStream_t stream) {
    (void)in_sizes; (void)n_in; (void)out_size; (void)ws_size;

    const float* x    = (const float*)d_in[0];
    const float* w1   = (const float*)d_in[1];
    const float* cb1  = (const float*)d_in[2];
    const float* g1   = (const float*)d_in[3];
    const float* bb1  = (const float*)d_in[4];
    const float* m1   = (const float*)d_in[5];
    const float* v1   = (const float*)d_in[6];
    const float* w2   = (const float*)d_in[7];
    const float* cb2  = (const float*)d_in[8];
    const float* g2   = (const float*)d_in[9];
    const float* bb2  = (const float*)d_in[10];
    const float* m2   = (const float*)d_in[11];
    const float* v2   = (const float*)d_in[12];
    const float* w3   = (const float*)d_in[13];
    const float* cb3  = (const float*)d_in[14];
    const float* g3   = (const float*)d_in[15];
    const float* bb3  = (const float*)d_in[16];
    const float* m3   = (const float*)d_in[17];
    const float* v3   = (const float*)d_in[18];
    const float* wb   = (const float*)d_in[19];
    const float* bbas = (const float*)d_in[20];
    const float* cwh  = (const float*)d_in[21];
    const float* cww  = (const float*)d_in[22];
    const float* fc1w = (const float*)d_in[23];
    const float* fc1b = (const float*)d_in[24];
    const float* fc2w = (const float*)d_in[25];
    const float* fc2b = (const float*)d_in[26];
    float* out = (float*)d_out;   // (64, 10) fp32

    size_t off = 0;
    auto alloc = [&](size_t bytes) -> void* {
        void* p = (char*)d_ws + off;
        off += (bytes + 255) & ~(size_t)255;
        return p;
    };

    float* sc1 = (float*)alloc(32 * 4);   float* bi1 = (float*)alloc(32 * 4);
    float* sc2 = (float*)alloc(64 * 4);   float* bi2 = (float*)alloc(64 * 4);
    float* sc3 = (float*)alloc(96 * 4);   float* bi3 = (float*)alloc(96 * 4);
    f16* w2h  = (f16*)alloc((size_t)64 * 288 * 2);
    f16* w3h  = (f16*)alloc((size_t)96 * 576 * 2);
    f16* wbh  = (f16*)alloc((size_t)512 * 864 * 2);
    f16* fc1h = (f16*)alloc((size_t)512 * 32768 * 2);
    f16* fc2h = (f16*)alloc((size_t)16 * 512 * 2);     // zero-padded to 16 rows
    f16* act1 = (f16*)alloc((size_t)64 * 32 * 64 * 64 * 2);
    f16* pre2 = (f16*)alloc((size_t)64 * 64 * 64 * 64 * 2);
    f16* act2 = (f16*)alloc((size_t)64 * 64 * 32 * 32 * 2);
    f16* pre3 = (f16*)alloc((size_t)64 * 96 * 32 * 32 * 2);
    f16* act3 = (f16*)alloc((size_t)64 * 96 * 16 * 16 * 2);
    float* ybasis = (float*)alloc((size_t)64 * 512 * 256 * 4);
    float* cwbuf  = (float*)alloc(256 * 4 * 4);
    f16* fcin = (f16*)alloc((size_t)64 * 32768 * 2);
    f16* fc1a = (f16*)alloc((size_t)64 * 512 * 2);

    // 1) fold BN into per-channel scale/bias (conv bias included)
    fold_bn_kernel<<<1, 128, 0, stream>>>(cb1, g1, bb1, m1, v1, sc1, bi1, 32);
    fold_bn_kernel<<<1, 128, 0, stream>>>(cb2, g2, bb2, m2, v2, sc2, bi2, 64);
    fold_bn_kernel<<<1, 128, 0, stream>>>(cb3, g3, bb3, m3, v3, sc3, bi3, 96);

    // 2) convert GEMM weights to f16 (FC2 zero-padded to 16 rows first)
    f32_to_f16_kernel<<<(64 * 288 + 255) / 256, 256, 0, stream>>>(w2, w2h, (size_t)64 * 288);
    f32_to_f16_kernel<<<(96 * 576 + 255) / 256, 256, 0, stream>>>(w3, w3h, (size_t)96 * 576);
    f32_to_f16_kernel<<<(512 * 864 + 255) / 256, 256, 0, stream>>>(wb, wbh, (size_t)512 * 864);
    f32_to_f16_kernel<<<(512 * 32768) / 256, 256, 0, stream>>>(fc1w, fc1h, (size_t)512 * 32768);
    zero_f16_kernel<<<(16 * 512 + 255) / 256, 256, 0, stream>>>(fc2h, 16 * 512);
    f32_to_f16_kernel<<<(10 * 512 + 255) / 256, 256, 0, stream>>>(fc2w, fc2h, (size_t)10 * 512);

    // 3) block 1: direct conv + BN + ReLU + pool fused
    conv1_fused_kernel<<<32768, 256, 0, stream>>>(x, w1, sc1, bi1, act1);

    // 4) block 2: WMMA implicit-GEMM conv + fused BN/ReLU, then pool
    conv_wmma_kernel<32, 64, 64, 64><<<dim3(128, 64), 256, 0, stream>>>(
        act1, w2h, sc2, bi2, pre2, nullptr, 0);
    pool2x2_kernel<<<(64 * 64 * 32 * 32) / 256, 256, 0, stream>>>(pre2, act2, 64, 64,
                                                                  64 * 64 * 32 * 32);

    // 5) block 3
    conv_wmma_kernel<64, 96, 32, 32><<<dim3(32, 64), 256, 0, stream>>>(
        act2, w3h, sc3, bi3, pre3, nullptr, 0);
    pool2x2_kernel<<<(64 * 96 * 16 * 16) / 256, 256, 0, stream>>>(pre3, act3, 32, 32,
                                                                  64 * 96 * 16 * 16);

    // 6) LRLC basis conv (raw f32 GEMM output), combining weights, combine
    conv_wmma_kernel<96, 512, 16, 16><<<dim3(8, 64), 256, 0, stream>>>(
        act3, wbh, nullptr, nullptr, nullptr, ybasis, 1);
    cw_softmax_kernel<<<1, 256, 0, stream>>>(cwh, cww, cwbuf);
    lrlc_combine_kernel<<<(64 * 128 * 256) / 256, 256, 0, stream>>>(ybasis, cwbuf, bbas, fcin);

    // 7) FC1 (relu) then FC2 -> fp32 logits
    // FC1: M=512, N=64, K=32768 -> 32*4 = 128 wave-tiles -> 16 blocks of 8 waves
    fc_wmma_kernel<<<16, 256, 0, stream>>>(fc1h, fcin, fc1b, fc1a, nullptr,
                                           512, 512, 64, 32768, 1, 512);
    // FC2: M=16 (10 valid), N=64, K=512 -> 4 wave-tiles -> 1 block
    fc_wmma_kernel<<<1, 256, 0, stream>>>(fc2h, fc1a, fc2b, nullptr, out,
                                          16, 10, 64, 512, 0, 10);
}
